// AttentionHead_21371757265260
// MI455X (gfx1250) — matrix operand order repaired and checked by reading, not compiled, not run on hardware
//
#include <hip/hip_runtime.h>
#include <hip/hip_bf16.h>

#define BB    8
#define SS    2048
#define DIN_  1024
#define DOUT_ 128

static constexpr float NEGV  = -1.0e9f;
static constexpr float SCALE = 0.022097086912079608f; // 1/sqrt(2048)

typedef __attribute__((ext_vector_type(16))) __bf16  v16bf;
typedef __attribute__((ext_vector_type(8)))  float   v8f;
typedef __attribute__((ext_vector_type(4)))  unsigned int v4u;

union BF16x16 {
    v16bf v;
    v4u   q[2];
    unsigned short u[16];
};
union BF16x8 {                 // 8 bf16 = one 16-byte store
    v4u q;
    unsigned short u[8];
};

__device__ __forceinline__ unsigned short f2bf(float f) {
    __bf16 h = (__bf16)f;      // lowers to v_cvt_pk_bf16_f32 when paired
    return __builtin_bit_cast(unsigned short, h);
}

// workspace layout (ushort elements)
#define QOFF   ((size_t)0)
#define KOFF   ((size_t)BB * SS * DOUT_)            // 2M
#define VOFF   ((size_t)2 * BB * SS * DOUT_)        // 4M  (swizzled)
#define WOFF   ((size_t)3 * BB * SS * DOUT_)        // 6M  (swizzled weights)

// ---------------------------------------------------------------------------
// Kernel 0: convert + swizzle weights into WMMA B-fragment order.
//   Wswz[wsel][kc][t][lane][j] = bf16( W[(kc*32 + 8*half + j + (j&8))*128 + t*16+l16] )
// grid = (32 kc, 24 = wsel*8+t), block = 32.  768 KB total, L2-resident.
// ---------------------------------------------------------------------------
__global__ __launch_bounds__(32)
void swizzle_w(const float* __restrict__ Wq, const float* __restrict__ Wk,
               const float* __restrict__ Wv, unsigned short* __restrict__ ws)
{
    const int kc   = blockIdx.x;            // 0..31 (k chunk of 32)
    const int wsel = blockIdx.y >> 3;       // 0..2
    const int t    = blockIdx.y & 7;        // 0..7 (N tile)
    const int lane = threadIdx.x & 31;
    const int half = lane >> 4;
    const int l16  = lane & 15;

    const float* W = (wsel == 0) ? Wq : (wsel == 1) ? Wk : Wv;
    unsigned short* dst =
        ws + WOFF + ((((size_t)wsel * 32 + kc) * 8 + t) * 32 + lane) * 16;

    BF16x16 o;
#pragma unroll
    for (int j = 0; j < 16; ++j)
        o.u[j] = f2bf(W[(size_t)(kc * 32 + 8 * half + j + (j & 8)) * DOUT_ + t * 16 + l16]);
    *(v4u*)(dst)     = o.q[0];
    *(v4u*)(dst + 8) = o.q[1];
}

// ---------------------------------------------------------------------------
// Kernel 1: fused QKV projection.  One wave = 16-row x 128-col stripe.
// A fragment: contiguous fp32 rows of X.  B fragment: 2 x b128 from Wswz.
// Q/K written row-major bf16; V written swizzled for kernel 2's B-fragments.
// grid = (B*S/16, 3), block = 32.
// ---------------------------------------------------------------------------
__global__ __launch_bounds__(32)
void qkv_proj_bf16_wmma(const float* __restrict__ X,
                        const float* __restrict__ bq,
                        const float* __restrict__ bk,
                        const float* __restrict__ bv,
                        unsigned short* __restrict__ ws)
{
    const int mtile = blockIdx.x;           // 0..1023
    const int wsel  = blockIdx.y;           // 0=Q 1=K 2=V

    const float* bias = (wsel == 0) ? bq : (wsel == 1) ? bk : bv;
    const unsigned short* Wswz = ws + WOFF + (size_t)wsel * 32 * 8 * 32 * 16;

    const int lane = threadIdx.x & 31;
    const int half = lane >> 4;
    const int l16  = lane & 15;

    const int rowA = mtile * 16 + l16;

    v8f acc[8];                             // bias folded into C init
#pragma unroll
    for (int t = 0; t < 8; ++t) {
        const float bc = bias[t * 16 + l16];
#pragma unroll
        for (int i = 0; i < 8; ++i) acc[t][i] = bc;
    }

    for (int k0 = 0; k0 < DIN_; k0 += 32) {
        BF16x16 a;                          // A: lane<16 K={0..7,16..23}
        const int ka = k0 + 8 * half;
        const float* xrow = X + (size_t)rowA * DIN_;
#pragma unroll
        for (int j = 0; j < 8; ++j) a.u[j]     = f2bf(xrow[ka + j]);
#pragma unroll
        for (int j = 0; j < 8; ++j) a.u[8 + j] = f2bf(xrow[ka + 16 + j]);

        const unsigned short* wbase =
            Wswz + (((size_t)(k0 >> 5) * 8) * 32 + lane) * 16;
#pragma unroll
        for (int t = 0; t < 8; ++t) {
            BF16x16 bm;                     // pre-swizzled: contiguous
            const unsigned short* wp = wbase + (size_t)t * 32 * 16;
            bm.q[0] = *(const v4u*)(wp);
            bm.q[1] = *(const v4u*)(wp + 8);
            acc[t] = __builtin_amdgcn_wmma_f32_16x16x32_bf16(
                         false, a.v, false, bm.v, (short)0, acc[t], false, false);
        }
    }

    if (wsel < 2) {
        // Q/K: row-major bf16 (attention reads contiguous 16B runs per row)
        unsigned short* out = ws + ((wsel == 0) ? QOFF : KOFF);
#pragma unroll
        for (int t = 0; t < 8; ++t)
#pragma unroll
            for (int i = 0; i < 8; ++i) {
                const int row = mtile * 16 + i + 8 * half;
                out[(size_t)row * DOUT_ + t * 16 + l16] = f2bf(acc[t][i]);
            }
    } else {
        // V: swizzled Vswz[mtile][t][l16][row16] -> one b128 store per tile
        unsigned short* out = ws + VOFF;
#pragma unroll
        for (int t = 0; t < 8; ++t) {
            BF16x8 p;
#pragma unroll
            for (int i = 0; i < 8; ++i) p.u[i] = f2bf(acc[t][i]);
            *(v4u*)(out + (((size_t)mtile * 8 + t) * 16 + l16) * 16 + 8 * half)
                = p.q;
        }
    }
}

// ---------------------------------------------------------------------------
// Kernel 2: flash attention, one wave per (batch, 16-row Q tile).
// 32 keys/iter: two score tiles, full-K (16x32)x(32x128) P*V.
// V B-fragments are 2 x b128 from the swizzled layout.
// grid = (S/16, B), block = 32.
// ---------------------------------------------------------------------------
__global__ __launch_bounds__(32)
void flash_attn_wmma(const unsigned short* __restrict__ ws,
                     const long long* __restrict__ amask,
                     float* __restrict__ out)
{
    const int qt   = blockIdx.x;            // 0..127
    const int b    = blockIdx.y;            // 0..7
    const int lane = threadIdx.x & 31;
    const int half = lane >> 4;
    const int l16  = lane & 15;

    const unsigned short* Q   = ws + QOFF;
    const unsigned short* K   = ws + KOFF;
    const unsigned short* Vsw = ws + VOFF;

    __shared__ unsigned short Plds[16 * 32];   // 1 KB, one wave per block

    BF16x16 qa[4];                          // resident Q tile (16 x 128)
    const size_t qbase = (size_t)(b * SS + qt * 16 + l16) * DOUT_;
#pragma unroll
    for (int c = 0; c < 4; ++c) {
        const int off = c * 32 + 16 * half;
        qa[c].q[0] = *(const v4u*)(Q + qbase + off);
        qa[c].q[1] = *(const v4u*)(Q + qbase + off + 8);
    }

    float m[8], l[8];
    v8f   o[8];
#pragma unroll
    for (int i = 0; i < 8; ++i) { m[i] = -3.0e38f; l[i] = 0.0f; }
#pragma unroll
    for (int t = 0; t < 8; ++t) o[t] = v8f{};

    for (int kt = 0; kt < SS / 32; ++kt) {
        const int   kpos0  = kt * 32 + l16;
        const int   kpos1  = kpos0 + 16;
        const size_t kbase0 = (size_t)(b * SS + kpos0) * DOUT_;
        const size_t kbase1 = (size_t)(b * SS + kpos1) * DOUT_;

        if (kt + 1 < SS / 32) {             // gfx1250 global_prefetch path
            __builtin_prefetch(K + kbase0 + (size_t)DOUT_ * 32, 0, 0);
            __builtin_prefetch(Vsw + ((size_t)(b * 128 + kt * 2 + 2) * 8) * 256, 0, 0);
        }

        // ---- two score tiles over 32 keys ----
        v8f s0 = {}, s1 = {};
#pragma unroll
        for (int c = 0; c < 4; ++c) {
            const int off = c * 32 + 16 * half;
            BF16x16 kb0, kb1;
            kb0.q[0] = *(const v4u*)(K + kbase0 + off);
            kb0.q[1] = *(const v4u*)(K + kbase0 + off + 8);
            kb1.q[0] = *(const v4u*)(K + kbase1 + off);
            kb1.q[1] = *(const v4u*)(K + kbase1 + off + 8);
            s0 = __builtin_amdgcn_wmma_f32_16x16x32_bf16(
                     false, qa[c].v, false, kb0.v, (short)0, s0, false, false);
            s1 = __builtin_amdgcn_wmma_f32_16x16x32_bf16(
                     false, qa[c].v, false, kb1.v, (short)0, s1, false, false);
        }

        // ---- scale + mask (per key column == per lane) ----
        const long long mv0 = amask[(size_t)b * SS + kpos0];
        const long long mv1 = amask[(size_t)b * SS + kpos1];
#pragma unroll
        for (int i = 0; i < 8; ++i) {
            s0[i] = (mv0 == 0) ? NEGV : s0[i] * SCALE;
            s1[i] = (mv1 == 0) ? NEGV : s1[i] * SCALE;
        }

        // ---- row max over 32 columns ----
        float mx[8];
#pragma unroll
        for (int i = 0; i < 8; ++i) mx[i] = fmaxf(s0[i], s1[i]);
#pragma unroll
        for (int off = 1; off < 16; off <<= 1)
#pragma unroll
            for (int i = 0; i < 8; ++i)
                mx[i] = fmaxf(mx[i], __shfl_xor(mx[i], off, 32));

        // ---- online softmax ----
        float alpha[8];
#pragma unroll
        for (int i = 0; i < 8; ++i) {
            const float mn = fmaxf(m[i], mx[i]);
            alpha[i] = __expf(m[i] - mn);
            m[i] = mn;
            s0[i] = __expf(s0[i] - mn);
            s1[i] = __expf(s1[i] - mn);
        }
        float ps[8];
#pragma unroll
        for (int i = 0; i < 8; ++i) ps[i] = s0[i] + s1[i];
#pragma unroll
        for (int off = 1; off < 16; off <<= 1)
#pragma unroll
            for (int i = 0; i < 8; ++i)
                ps[i] += __shfl_xor(ps[i], off, 32);
#pragma unroll
        for (int i = 0; i < 8; ++i) l[i] = l[i] * alpha[i] + ps[i];
#pragma unroll
        for (int t = 0; t < 8; ++t)
#pragma unroll
            for (int i = 0; i < 8; ++i) o[t][i] *= alpha[i];

        // ---- C-layout -> A-layout for P (16x32) via LDS ----
#pragma unroll
        for (int i = 0; i < 8; ++i) {
            Plds[(i + 8 * half) * 32 + l16]      = f2bf(s0[i]);
            Plds[(i + 8 * half) * 32 + 16 + l16] = f2bf(s1[i]);
        }
        __syncthreads();

        BF16x16 pa;
#pragma unroll
        for (int j = 0; j < 8; ++j) {
            pa.u[j]     = Plds[l16 * 32 + 16 * half + j];
            pa.u[8 + j] = Plds[l16 * 32 + 16 * half + 8 + j];
        }

        // ---- O += P(16x32) * V(32x128), V fragments contiguous ----
        // Vswz tile index: b*128 + kt*2 + half holds this lane-half's 16 rows
        const size_t vtile = (size_t)(b * 128 + kt * 2 + half);
#pragma unroll
        for (int t = 0; t < 8; ++t) {
            const unsigned short* vp =
                Vsw + ((vtile * 8 + t) * 16 + l16) * 16;
            BF16x16 vb;
            vb.q[0] = *(const v4u*)(vp);
            vb.q[1] = *(const v4u*)(vp + 8);
            o[t] = __builtin_amdgcn_wmma_f32_16x16x32_bf16(
                       false, pa.v, false, vb.v, (short)0, o[t], false, false);
        }
        __syncthreads();
    }

    // ---- normalize and store fp32 output ----
#pragma unroll
    for (int i = 0; i < 8; ++i) {
        const float inv = 1.0f / l[i];
        const size_t row = (size_t)(b * SS + qt * 16 + i + 8 * half) * DOUT_;
#pragma unroll
        for (int t = 0; t < 8; ++t)
            out[row + t * 16 + l16] = o[t][i] * inv;
    }
}

// ---------------------------------------------------------------------------
extern "C" void kernel_launch(void* const* d_in, const int* in_sizes, int n_in,
                              void* d_out, int out_size, void* d_ws, size_t ws_size,
                              hipStream_t stream) {
    const float*     X   = (const float*)d_in[0];
    const long long* msk = (const long long*)d_in[1];
    const float*     Wq  = (const float*)d_in[2];
    const float*     bq  = (const float*)d_in[3];
    const float*     Wk  = (const float*)d_in[4];
    const float*     bk  = (const float*)d_in[5];
    const float*     Wv  = (const float*)d_in[6];
    const float*     bv  = (const float*)d_in[7];
    float*           out = (float*)d_out;

    unsigned short* ws = (unsigned short*)d_ws;   // ~13.2 MB used

    dim3 g0(32, 24);               // 32 k-chunks x (3 weights * 8 N-tiles)
    swizzle_w<<<g0, 32, 0, stream>>>(Wq, Wk, Wv, ws);

    dim3 g1(BB * SS / 16, 3);      // 1024 M-tiles x {Q,K,V}
    qkv_proj_bf16_wmma<<<g1, 32, 0, stream>>>(X, bq, bk, bv, ws);

    dim3 g2(SS / 16, BB);          // 128 Q-tiles x 8 batches
    flash_attn_wmma<<<g2, 32, 0, stream>>>(ws, msk, out);
}
